// AttentionLayer_16595753631878
// MI455X (gfx1250) — matrix operand order repaired and checked by reading, not compiled, not run on hardware
//
#include <hip/hip_runtime.h>
#include <stdint.h>

// ---------------------------------------------------------------------------
// Types for CDNA5 WMMA (wave32, 16x16x32 bf16 -> f32)
// ---------------------------------------------------------------------------
typedef __attribute__((ext_vector_type(16))) __bf16         v16bf;
typedef __attribute__((ext_vector_type(16))) unsigned short v16us;
typedef __attribute__((ext_vector_type(8)))  float          v8f;
typedef __attribute__((ext_vector_type(4)))  unsigned int   v4u;

union Frag16 {
    v16bf bf;
    v16us us;
    v4u   q[2];
};

__device__ __forceinline__ unsigned short f2bf_u(float f) {
    unsigned u = __builtin_bit_cast(unsigned, f);
    unsigned r = u + 0x7FFFu + ((u >> 16) & 1u);   // round to nearest even
    return (unsigned short)(r >> 16);
}

__device__ __forceinline__ v8f wmma_bf16(const Frag16& a, const Frag16& b, v8f c) {
    return __builtin_amdgcn_wmma_f32_16x16x32_bf16(
        /*neg_a=*/false, a.bf, /*neg_b=*/false, b.bf,
        /*c_mod=*/(short)0, c, /*reuse_a=*/false, /*reuse_b=*/false);
}

// ---------------------------------------------------------------------------
// CDNA5 async global->LDS copy (bypasses VGPRs, tracked on ASYNCcnt).
// LDS generic pointers on AMDGPU carry the LDS byte offset in the low 32 bits
// (ISA 10.2: LDS aperture address truncates to 32 bits), so we pass that.
// ---------------------------------------------------------------------------
__device__ __forceinline__ void async_load_b128(void* lds_ptr, const void* gaddr) {
    unsigned lds_off = (unsigned)(uintptr_t)lds_ptr;
    asm volatile("global_load_async_to_lds_b128 %0, %1, off"
                 :: "v"(lds_off), "v"(gaddr)
                 : "memory");
}
__device__ __forceinline__ void async_wait0() {
    asm volatile("s_wait_asynccnt 0x0" ::: "memory");
}

// ---------------------------------------------------------------------------
// Problem constants
// ---------------------------------------------------------------------------
#define BB   4
#define SS   2048
#define HH   16
#define DD   128
#define EE   2048          // H*D
#define FF   8192          // 4*E
#define BS   (BB*SS)       // 8192
#define EPSV 1e-5f

// ---------------------------------------------------------------------------
// fp32 -> bf16 convert
// ---------------------------------------------------------------------------
__global__ __launch_bounds__(256) void cvt_bf16(const float* __restrict__ src,
                                                unsigned short* __restrict__ dst,
                                                size_t n) {
    size_t stride = (size_t)gridDim.x * blockDim.x;
    for (size_t i = (size_t)blockIdx.x * blockDim.x + threadIdx.x; i < n; i += stride)
        dst[i] = f2bf_u(src[i]);
}

// ---------------------------------------------------------------------------
// GEMM: Out[M,N] = A[M,K](bf16) * W[N,K](bf16)^T + bias[N]  (opt. ReLU)
// Block: 256 threads = 8 waves. Block tile 128x128, wave tile 32x64
// (2x4 WMMA frags). Double-buffered LDS filled by async-to-LDS loads.
// ---------------------------------------------------------------------------
template <bool RELU, bool OUT_BF16>
__global__ __launch_bounds__(256)
void gemm_bf16(const unsigned short* __restrict__ A,
               const unsigned short* __restrict__ W,
               const float* __restrict__ bias,
               void* __restrict__ Out,
               int M, int N, int K) {
    constexpr int BM = 128, BN = 128, BK = 32, PAD = 8;
    __shared__ unsigned short As[2][BM][BK + PAD];
    __shared__ unsigned short Bs[2][BN][BK + PAD];

    const int mbase = blockIdx.y * BM;
    const int nbase = blockIdx.x * BN;
    const int t     = threadIdx.x;
    const int wave  = t >> 5;
    const int lane  = t & 31;
    const int wm    = wave >> 1;       // 0..3 -> 32-row strip
    const int wn    = wave & 1;        // 0..1 -> 64-col strip
    const int half  = lane >> 4;       // 0/1
    const int l16   = lane & 15;

    // Each thread stages 32B of A and 32B of W per K-step via async loads.
    const int srow = t >> 1, sseg = t & 1;

    auto stage = [&](int k0, int p) {
        const unsigned short* ga = A + (size_t)(mbase + srow) * K + k0 + sseg * 16;
        unsigned short* la = &As[p][srow][sseg * 16];
        async_load_b128(la,     ga);
        async_load_b128(la + 8, ga + 8);
        const unsigned short* gw = W + (size_t)(nbase + srow) * K + k0 + sseg * 16;
        unsigned short* lb = &Bs[p][srow][sseg * 16];
        async_load_b128(lb,     gw);
        async_load_b128(lb + 8, gw + 8);
    };

    v8f acc[2][4] = {};

    const int nsteps = K / BK;
    stage(0, 0);
    async_wait0();
    __syncthreads();

    for (int s = 0; s < nsteps; s++) {
        const int p = s & 1;
        if (s + 1 < nsteps) stage((s + 1) * BK, p ^ 1);   // prefetch next tile

        Frag16 af[2], bfr[4];
#pragma unroll
        for (int i = 0; i < 2; i++) {
            int m = wm * 32 + i * 16 + l16;
            af[i].q[0] = *reinterpret_cast<const v4u*>(&As[p][m][half * 8]);
            af[i].q[1] = *reinterpret_cast<const v4u*>(&As[p][m][16 + half * 8]);
        }
#pragma unroll
        for (int j = 0; j < 4; j++) {
            int n = wn * 64 + j * 16 + l16;
            bfr[j].q[0] = *reinterpret_cast<const v4u*>(&Bs[p][n][half * 8]);
            bfr[j].q[1] = *reinterpret_cast<const v4u*>(&Bs[p][n][16 + half * 8]);
        }
#pragma unroll
        for (int i = 0; i < 2; i++)
#pragma unroll
            for (int j = 0; j < 4; j++)
                acc[i][j] = wmma_bf16(af[i], bfr[j], acc[i][j]);

        async_wait0();          // own prefetch landed
        __syncthreads();        // everyone done reading buf p / writing p^1
    }

    // Epilogue
#pragma unroll
    for (int i = 0; i < 2; i++)
#pragma unroll
        for (int j = 0; j < 4; j++) {
            int col = nbase + wn * 64 + j * 16 + l16;
            float bv = bias[col];
#pragma unroll
            for (int r = 0; r < 8; r++) {
                int row = mbase + wm * 32 + i * 16 + r + 8 * half;
                float v = acc[i][j][r] + bv;
                if (RELU) v = fmaxf(v, 0.f);
                if (OUT_BF16)
                    ((unsigned short*)Out)[(size_t)row * N + col] = f2bf_u(v);
                else
                    ((float*)Out)[(size_t)row * N + col] = v;
            }
        }
}

// ---------------------------------------------------------------------------
// Attention: per block handles (b, h, 16-row q tile).
//   pass1: logits[16][S] = Q * K^T into LDS (WMMA)
//   fp32 softmax, pass2: O = P * V (WMMA, async double-buffered V), bf16 out
// ---------------------------------------------------------------------------
__global__ __launch_bounds__(256)
void attn_kernel(const unsigned short* __restrict__ Q,
                 const unsigned short* __restrict__ Kb,
                 const unsigned short* __restrict__ V,
                 unsigned short* __restrict__ O) {
    __shared__ unsigned short Qs[16][DD + 8];
    __shared__ unsigned short Vs[2][32][DD + 8];
    __shared__ float Lg[16][SS];

    const int nqt = SS / 16;                    // 128
    const int qt  = blockIdx.x & (nqt - 1);
    const int bh  = blockIdx.x >> 7;
    const int h   = bh & (HH - 1);
    const int b   = bh >> 4;

    const int t    = threadIdx.x;
    const int wave = t >> 5;
    const int lane = t & 31;
    const int half = lane >> 4;
    const int l16  = lane & 15;

    // Load Q tile (16 x 128 bf16), coalesced
    {
        int row = t >> 4, seg = t & 15;
        const v4u* src = reinterpret_cast<const v4u*>(
            Q + ((size_t)(b * SS + qt * 16 + row) * EE + h * DD + seg * 8));
        *reinterpret_cast<v4u*>(&Qs[row][seg * 8]) = src[0];
    }
    __syncthreads();

    // A fragments for this wave (reused for every K tile)
    Frag16 af[4];
#pragma unroll
    for (int c = 0; c < 4; c++) {
        af[c].q[0] = *reinterpret_cast<const v4u*>(&Qs[l16][c * 32 + half * 8]);
        af[c].q[1] = *reinterpret_cast<const v4u*>(&Qs[l16][c * 32 + 16 + half * 8]);
    }

    // Pass 1: each wave computes t-tiles tt = wave, wave+8, ... (K stays in L2)
    for (int tt = wave; tt < nqt; tt += 8) {
        const unsigned short* krow =
            Kb + ((size_t)(b * SS + tt * 16 + l16) * EE + h * DD);
        v8f cc = {};
#pragma unroll
        for (int c = 0; c < 4; c++) {
            Frag16 bfr;
            const v4u* kp = reinterpret_cast<const v4u*>(krow + c * 32 + half * 8);
            bfr.q[0] = kp[0];
            bfr.q[1] = kp[2];
            cc = wmma_bf16(af[c], bfr, cc);
        }
#pragma unroll
        for (int r = 0; r < 8; r++)
            Lg[r + 8 * half][tt * 16 + l16] = cc[r];
    }
    __syncthreads();

    // Softmax over t (scaled by 1/sqrt(D)); 16 threads cooperate per row
    {
        const float scale = rsqrtf((float)DD);
        int row = t >> 4, i = t & 15;
        float mx = -3.4e38f;
        for (int c = i; c < SS; c += 16) mx = fmaxf(mx, Lg[row][c]);
#pragma unroll
        for (int off = 8; off >= 1; off >>= 1)
            mx = fmaxf(mx, __shfl_xor(mx, off, 16));
        float sum = 0.f;
        for (int c = i; c < SS; c += 16) {
            float p = __expf((Lg[row][c] - mx) * scale);
            Lg[row][c] = p;
            sum += p;
        }
#pragma unroll
        for (int off = 8; off >= 1; off >>= 1)
            sum += __shfl_xor(sum, off, 16);
        float inv = 1.f / sum;
        for (int c = i; c < SS; c += 16) Lg[row][c] *= inv;
    }

    // Pass 2: O(16x128) = P(16xS) * V(SxD); each wave owns 16 columns of D.
    // V tiles (32x128 bf16) double-buffered via async-to-LDS.
    const int vrow = t >> 3, vseg = t & 7;
    auto stageV = [&](int kc, int p) {
        const unsigned short* g =
            V + ((size_t)(b * SS + kc * 32 + vrow) * EE + h * DD + vseg * 16);
        unsigned short* l = &Vs[p][vrow][vseg * 16];
        async_load_b128(l,     g);
        async_load_b128(l + 8, g + 8);
    };

    __syncthreads();            // softmax results visible to all
    stageV(0, 0);
    async_wait0();
    __syncthreads();

    const int dbase = wave * 16;
    v8f acc = {};
    for (int kc = 0; kc < SS / 32; kc++) {
        const int p = kc & 1;
        if (kc + 1 < SS / 32) stageV(kc + 1, p ^ 1);   // prefetch next V tile

        // A fragment: probs row l16, K-chunk kc (fp32 -> bf16)
        Frag16 pa;
#pragma unroll
        for (int j = 0; j < 8; j++)
            pa.us[j] = f2bf_u(Lg[l16][kc * 32 + half * 8 + j]);
#pragma unroll
        for (int j = 0; j < 8; j++)
            pa.us[8 + j] = f2bf_u(Lg[l16][kc * 32 + 16 + half * 8 + j]);

        // B fragment: V^T chunk — column (dbase+l16), rows half*8+j (j<8), 16+half*8+(j-8) (j>=8)
        Frag16 vb;
#pragma unroll
        for (int j = 0; j < 16; j++)
            vb.us[j] = Vs[p][(j < 8 ? half * 8 + j : 16 + half * 8 + (j - 8))][dbase + l16];

        acc = wmma_bf16(pa, vb, acc);

        async_wait0();
        __syncthreads();
    }

#pragma unroll
    for (int r = 0; r < 8; r++) {
        int row = qt * 16 + r + 8 * half;
        O[(size_t)(b * SS + row) * EE + h * DD + dbase + l16] = f2bf_u(acc[r]);
    }
}

// ---------------------------------------------------------------------------
// LayerNorm over dim=1 (sequence), ddof=1, fused residual add.
// ---------------------------------------------------------------------------
__global__ __launch_bounds__(256)
void ln_partial(const float* __restrict__ a, const float* __restrict__ b,
                float* __restrict__ psum, float* __restrict__ psq) {
    const int SPLIT = 16;
    int e  = blockIdx.x * blockDim.x + threadIdx.x;   // 0..E-1
    int bb = blockIdx.y;
    int s0 = blockIdx.z * (SS / SPLIT);
    float s1 = 0.f, s2 = 0.f;
    for (int s = s0; s < s0 + SS / SPLIT; ++s) {
        size_t idx = ((size_t)bb * SS + s) * EE + e;
        float z = a[idx] + b[idx];
        s1 += z;
        s2 += z * z;
    }
    atomicAdd(&psum[bb * EE + e], s1);
    atomicAdd(&psq[bb * EE + e], s2);
}

template <bool WRITE_BF16>
__global__ __launch_bounds__(256)
void ln_final(const float* __restrict__ a, const float* __restrict__ b,
              const float* __restrict__ psum, const float* __restrict__ psq,
              float* __restrict__ outf, unsigned short* __restrict__ outb) {
    size_t idx = (size_t)blockIdx.x * blockDim.x + threadIdx.x;
    if (idx >= (size_t)BS * EE) return;
    int e  = (int)(idx % EE);
    int bb = (int)(idx / ((size_t)SS * EE));
    float z    = a[idx] + b[idx];
    float mean = psum[bb * EE + e] * (1.f / SS);
    float var  = (psq[bb * EE + e] - (float)SS * mean * mean) * (1.f / (SS - 1));
    float y    = (z - mean) * rsqrtf(var + EPSV);
    outf[idx] = y;
    if (WRITE_BF16) outb[idx] = f2bf_u(y);
}

// ---------------------------------------------------------------------------
// Host-side orchestration
// ---------------------------------------------------------------------------
extern "C" void kernel_launch(void* const* d_in, const int* in_sizes, int n_in,
                              void* d_out, int out_size, void* d_ws, size_t ws_size,
                              hipStream_t stream) {
    const float* x  = (const float*)d_in[0];
    const float* Wq = (const float*)d_in[1];
    const float* bq = (const float*)d_in[2];
    const float* Wk = (const float*)d_in[3];
    const float* bk = (const float*)d_in[4];
    const float* Wv = (const float*)d_in[5];
    const float* bv = (const float*)d_in[6];
    const float* Wo = (const float*)d_in[7];
    const float* bo = (const float*)d_in[8];
    const float* W1 = (const float*)d_in[9];
    const float* b1 = (const float*)d_in[10];
    const float* W2 = (const float*)d_in[11];
    const float* b2 = (const float*)d_in[12];
    float* out = (float*)d_out;

    const size_t MB = 1024ull * 1024ull;
    char* ws = (char*)d_ws;
    unsigned short* xb   = (unsigned short*)(ws + 0 * MB);     // 32MB
    unsigned short* wqb  = (unsigned short*)(ws + 32 * MB);    // 8MB
    unsigned short* wkb  = (unsigned short*)(ws + 40 * MB);    // 8MB
    unsigned short* wvb  = (unsigned short*)(ws + 48 * MB);    // 8MB
    unsigned short* wob  = (unsigned short*)(ws + 56 * MB);    // 8MB
    unsigned short* w1b  = (unsigned short*)(ws + 64 * MB);    // 32MB
    unsigned short* w2b  = (unsigned short*)(ws + 96 * MB);    // 32MB
    unsigned short* qb   = (unsigned short*)(ws + 128 * MB);   // 32MB
    unsigned short* kb   = (unsigned short*)(ws + 160 * MB);   // 32MB
    unsigned short* vb   = (unsigned short*)(ws + 192 * MB);   // 32MB
    unsigned short* ob   = (unsigned short*)(ws + 224 * MB);   // 32MB
    unsigned short* hb   = (unsigned short*)(ws + 128 * MB);   // 128MB, overlays q/k/v/o
    float* scratchf      = (float*)(ws + 256 * MB);            // 64MB (attn, then ffn)
    float* x1f           = (float*)(ws + 320 * MB);            // 64MB
    unsigned short* x1b  = (unsigned short*)(ws + 384 * MB);   // 32MB
    float* stat_sum      = (float*)(ws + 416 * MB);            // 32KB
    float* stat_sq       = (float*)(ws + 416 * MB + 64 * 1024);

    // 1) convert activations + weights to bf16
    cvt_bf16<<<4096, 256, 0, stream>>>(x,  xb,  (size_t)BS * EE);
    cvt_bf16<<<4096, 256, 0, stream>>>(Wq, wqb, (size_t)EE * EE);
    cvt_bf16<<<4096, 256, 0, stream>>>(Wk, wkb, (size_t)EE * EE);
    cvt_bf16<<<4096, 256, 0, stream>>>(Wv, wvb, (size_t)EE * EE);
    cvt_bf16<<<4096, 256, 0, stream>>>(Wo, wob, (size_t)EE * EE);
    cvt_bf16<<<4096, 256, 0, stream>>>(W1, w1b, (size_t)FF * EE);
    cvt_bf16<<<4096, 256, 0, stream>>>(W2, w2b, (size_t)EE * FF);

    // 2) Q/K/V projections: [BS,E] x [E,E]^T -> bf16 [B,S,H,D]
    dim3 gProj(EE / 128, BS / 128);
    gemm_bf16<false, true><<<gProj, 256, 0, stream>>>(xb, wqb, bq, qb, BS, EE, EE);
    gemm_bf16<false, true><<<gProj, 256, 0, stream>>>(xb, wkb, bk, kb, BS, EE, EE);
    gemm_bf16<false, true><<<gProj, 256, 0, stream>>>(xb, wvb, bv, vb, BS, EE, EE);

    // 3) attention (softmax(QK^T/sqrt(D)) V), concat heads -> ob (bf16 [BS,E])
    attn_kernel<<<BB * HH * (SS / 16), 256, 0, stream>>>(qb, kb, vb, ob);

    // 4) output projection -> scratchf (fp32)
    gemm_bf16<false, false><<<gProj, 256, 0, stream>>>(ob, wob, bo, scratchf, BS, EE, EE);

    // 5) x1 = LayerNorm_dim1(attn + x)  -> x1f (fp32) + x1b (bf16)
    hipMemsetAsync(stat_sum, 0, (size_t)BB * EE * sizeof(float), stream);
    hipMemsetAsync(stat_sq,  0, (size_t)BB * EE * sizeof(float), stream);
    ln_partial<<<dim3(EE / 256, BB, 16), 256, 0, stream>>>(scratchf, x, stat_sum, stat_sq);
    ln_final<true><<<(unsigned)(((size_t)BS * EE) / 256), 256, 0, stream>>>(
        scratchf, x, stat_sum, stat_sq, x1f, x1b);

    // 6) FFN1: h = relu(x1 @ W1^T + b1) -> hb (bf16 [BS,4E])
    gemm_bf16<true, true><<<dim3(FF / 128, BS / 128), 256, 0, stream>>>(
        x1b, w1b, b1, hb, BS, FF, EE);

    // 7) FFN2: ffn = h @ W2^T + b2 -> scratchf (fp32)
    gemm_bf16<false, false><<<dim3(EE / 128, BS / 128), 256, 0, stream>>>(
        hb, w2b, b2, scratchf, BS, EE, FF);

    // 8) out = LayerNorm_dim1(ffn + x1)
    hipMemsetAsync(stat_sum, 0, (size_t)BB * EE * sizeof(float), stream);
    hipMemsetAsync(stat_sq,  0, (size_t)BB * EE * sizeof(float), stream);
    ln_partial<<<dim3(EE / 256, BB, 16), 256, 0, stream>>>(scratchf, x1f, stat_sum, stat_sq);
    ln_final<false><<<(unsigned)(((size_t)BS * EE) / 256), 256, 0, stream>>>(
        scratchf, x1f, stat_sum, stat_sq, out, nullptr);
}
